// SparseResBlock3d_9431748182657
// MI455X (gfx1250) — matrix operand order repaired and checked by reading, not compiled, not run on hardware
//
#include <hip/hip_runtime.h>
#include <hip/hip_bf16.h>

typedef __attribute__((ext_vector_type(16))) _Float16 v16h;
typedef __attribute__((ext_vector_type(8)))  _Float16 v8h;
typedef __attribute__((ext_vector_type(8)))  float    v8f;

#define KOFF 27
#define CCH  64
#define EMB  512
// packed weight fragments: [27][2 kchunk][4 cout-tile][32 lanes][16 halves]
#define PACK_ELEMS (KOFF * 2 * 4 * 32 * 16)

__device__ __forceinline__ float silu_f(float x) {
    return x * (1.0f / (1.0f + __expf(-x)));
}

__device__ __forceinline__ float wave_sum(float v) {
    #pragma unroll
    for (int m = 16; m >= 1; m >>= 1) v += __shfl_xor(v, m, 32);
    return v;
}

// ---------------------------------------------------------------------------
// Pack W [27,64,64] f32 -> f16 B-matrix fragments for V_WMMA_F32_16X16X32_F16.
// B (32x16): lane holds column n = lane&15, K = (lane>>4)*8 + j (j<8), 16 + (lane>>4)*8 + (j-8) (j>=8).
// ---------------------------------------------------------------------------
__global__ void pack_weights(const float* __restrict__ W, _Float16* __restrict__ P) {
    int tid = blockIdx.x * blockDim.x + threadIdx.x;
    if (tid >= PACK_ELEMS) return;
    int j    = tid & 15;
    int lane = (tid >> 4) & 31;
    int t    = (tid >> 9) & 3;
    int c    = (tid >> 11) & 1;
    int k    = tid >> 12;
    int n    = lane & 15;
    int K    = ((lane >> 4) << 3) + (j & 7) + ((j >> 3) << 4);
    int cin  = c * 32 + K;
    int cout = t * 16 + n;
    P[tid] = (_Float16)W[(k * CCH + cin) * CCH + cout];
}

// ---------------------------------------------------------------------------
// FiLM embedding MLP: ss[b, j] = sum_e silu(emb[b,e]) * emb_W[e,j] + emb_b[j]
// ---------------------------------------------------------------------------
__global__ void emb_mlp(const float* __restrict__ emb, const float* __restrict__ Wm,
                        const float* __restrict__ bm, float* __restrict__ ss) {
    int j = threadIdx.x;   // 0..127
    int b = blockIdx.x;    // 0..B-1
    float acc = bm[j];
    for (int e = 0; e < EMB; ++e) {
        acc += silu_f(emb[b * EMB + e]) * Wm[e * (2 * CCH) + j];
    }
    ss[b * (2 * CCH) + j] = acc;
}

// ---------------------------------------------------------------------------
// LayerNorm(gamma,beta) + SiLU, f32 -> f16, one wave32 per row, zero pad row N.
// ---------------------------------------------------------------------------
__global__ void ln1_silu(const float* __restrict__ x, const float* __restrict__ gamma,
                         const float* __restrict__ beta, _Float16* __restrict__ h, int N) {
    int row  = blockIdx.x * (blockDim.x >> 5) + (threadIdx.x >> 5);
    int lane = threadIdx.x & 31;
    if (row > N) return;
    int c0 = lane * 2;
    _Float16* hp = h + (size_t)row * CCH;
    if (row == N) { hp[c0] = (_Float16)0.f; hp[c0 + 1] = (_Float16)0.f; return; }
    const float* xp = x + (size_t)row * CCH;
    float x0 = xp[c0], x1 = xp[c0 + 1];
    float s  = wave_sum(x0 + x1);
    float sq = wave_sum(x0 * x0 + x1 * x1);
    float mean = s * (1.0f / CCH);
    float var  = sq * (1.0f / CCH) - mean * mean;
    float inv  = rsqrtf(var + 1e-6f);
    float y0 = silu_f((x0 - mean) * inv * gamma[c0]     + beta[c0]);
    float y1 = silu_f((x1 - mean) * inv * gamma[c0 + 1] + beta[c0 + 1]);
    hp[c0] = (_Float16)y0; hp[c0 + 1] = (_Float16)y1;
}

// ---------------------------------------------------------------------------
// LayerNorm (no affine) + FiLM + SiLU, f32 -> f16, zero pad row N.
// ---------------------------------------------------------------------------
__global__ void film_silu(const float* __restrict__ x, const float* __restrict__ ss,
                          const int* __restrict__ bidx, _Float16* __restrict__ h, int N) {
    int row  = blockIdx.x * (blockDim.x >> 5) + (threadIdx.x >> 5);
    int lane = threadIdx.x & 31;
    if (row > N) return;
    int c0 = lane * 2;
    _Float16* hp = h + (size_t)row * CCH;
    if (row == N) { hp[c0] = (_Float16)0.f; hp[c0 + 1] = (_Float16)0.f; return; }
    const float* xp = x + (size_t)row * CCH;
    float x0 = xp[c0], x1 = xp[c0 + 1];
    float s  = wave_sum(x0 + x1);
    float sq = wave_sum(x0 * x0 + x1 * x1);
    float mean = s * (1.0f / CCH);
    float var  = sq * (1.0f / CCH) - mean * mean;
    float inv  = rsqrtf(var + 1e-6f);
    int b = bidx[row];
    const float* sb = ss + b * (2 * CCH);
    float y0 = silu_f((x0 - mean) * inv * (1.0f + sb[c0])     + sb[CCH + c0]);
    float y1 = silu_f((x1 - mean) * inv * (1.0f + sb[c0 + 1]) + sb[CCH + c0 + 1]);
    hp[c0] = (_Float16)y0; hp[c0 + 1] = (_Float16)y1;
}

// ---------------------------------------------------------------------------
// Sparse conv: each wave32 computes a 16-voxel x 64-channel output tile.
// Gather-GEMM over 27 offsets, K chunked by 32 -> V_WMMA_F32_16X16X32_F16.
// A fragment (16x32 f16): lane m = lane&15, kbase = (lane>>4)*8,
//   halves 0..7  = K kbase..kbase+7, halves 8..15 = K kbase+16..kbase+23
//   -> two 16-byte loads from the gathered row.
// ---------------------------------------------------------------------------
template <bool RESID>
__global__ void spconv_wmma(const _Float16* __restrict__ h,   // [N+1, 64] f16 (row N = 0)
                            const _Float16* __restrict__ Wp,  // packed fragments
                            const float* __restrict__ bias,   // [64]
                            const int* __restrict__ nidx,     // [N, 27]
                            const float* __restrict__ resid,  // [N, 64] or null
                            float* __restrict__ out,          // [N, 64]
                            int Nrows) {
    const int lane = threadIdx.x & 31;
    const int wave = threadIdx.x >> 5;
    const int tile = blockIdx.x * (blockDim.x >> 5) + wave;
    const int row0 = tile * 16;
    if (row0 >= Nrows) return;                    // wave-uniform: EXEC stays all-1s

    const int m     = lane & 15;
    const int hi    = lane >> 4;
    const int kbase = hi * 8;
    const int myrow = row0 + m;

    v8f acc[4] = {};

    int idx = nidx[myrow * KOFF + 0];
    for (int k = 0; k < KOFF; ++k) {
        const _Float16* rp = h + (size_t)idx * CCH;
        if (k + 1 < KOFF) {                       // prefetch next gathered row
            idx = nidx[myrow * KOFF + k + 1];
            __builtin_prefetch(h + (size_t)idx * CCH, 0, 0);
        }
        #pragma unroll
        for (int c = 0; c < 2; ++c) {
            v16h a;
            *((v8h*)&a)     = *(const v8h*)(rp + c * 32 + kbase);
            *((v8h*)&a + 1) = *(const v8h*)(rp + c * 32 + kbase + 16);
            const _Float16* wp = Wp + (size_t)(((k * 2 + c) * 4) * 32 + lane) * 16;
            #pragma unroll
            for (int t = 0; t < 4; ++t) {
                v16h b = *(const v16h*)(wp + (size_t)t * 32 * 16);
                acc[t] = __builtin_amdgcn_wmma_f32_16x16x32_f16(
                             false, a, false, b, (short)0, acc[t], false, false);
            }
        }
    }

    const float bb0 = bias[m], bb1 = bias[16 + m], bb2 = bias[32 + m], bb3 = bias[48 + m];
    #pragma unroll
    for (int r = 0; r < 8; ++r) {
        const int orow = row0 + hi * 8 + r;       // C/D layout: VGPR r -> M = hi*8 + r
        float* op = out + (size_t)orow * CCH + m;
        float r0 = acc[0][r] + bb0;
        float r1 = acc[1][r] + bb1;
        float r2 = acc[2][r] + bb2;
        float r3 = acc[3][r] + bb3;
        if (RESID) {
            const float* sp = resid + (size_t)orow * CCH + m;
            r0 += sp[0]; r1 += sp[16]; r2 += sp[32]; r3 += sp[48];
        }
        op[0] = r0; op[16] = r1; op[32] = r2; op[48] = r3;
    }
}

extern "C" void kernel_launch(void* const* d_in, const int* in_sizes, int n_in,
                              void* d_out, int out_size, void* d_ws, size_t ws_size,
                              hipStream_t stream) {
    const float* feats  = (const float*)d_in[0];
    const float* emb    = (const float*)d_in[1];
    const float* gamma  = (const float*)d_in[2];
    const float* beta   = (const float*)d_in[3];
    const float* W1     = (const float*)d_in[4];
    const float* b1     = (const float*)d_in[5];
    const float* W2     = (const float*)d_in[6];
    const float* b2     = (const float*)d_in[7];
    const float* emb_W  = (const float*)d_in[8];
    const float* emb_b  = (const float*)d_in[9];
    const int*   nidx   = (const int*)d_in[10];
    const int*   bidx   = (const int*)d_in[11];
    float*       outp   = (float*)d_out;

    const int N = in_sizes[0] / CCH;        // 131072
    const int B = in_sizes[1] / EMB;        // 4

    auto align256 = [](size_t x) { return (x + 255) & ~(size_t)255; };
    char* w = (char*)d_ws;
    size_t szH  = (size_t)(N + 1) * CCH * sizeof(_Float16);
    size_t szP  = (size_t)PACK_ELEMS * sizeof(_Float16);
    size_t offH1 = 0;
    size_t offH2 = align256(offH1 + szH);
    size_t offP1 = align256(offH2 + szH);
    size_t offP2 = align256(offP1 + szP);
    size_t offSS = align256(offP2 + szP);
    size_t offC1 = align256(offSS + (size_t)B * 2 * CCH * sizeof(float));

    _Float16* h1  = (_Float16*)(w + offH1);
    _Float16* h2  = (_Float16*)(w + offH2);
    _Float16* P1  = (_Float16*)(w + offP1);
    _Float16* P2  = (_Float16*)(w + offP2);
    float*    ss  = (float*)(w + offSS);
    float*    c1o = (float*)(w + offC1);

    // 1) pack weights (f32 -> f16 WMMA B-fragments)
    {
        int thr = 256, blk = (PACK_ELEMS + thr - 1) / thr;
        pack_weights<<<blk, thr, 0, stream>>>(W1, P1);
        pack_weights<<<blk, thr, 0, stream>>>(W2, P2);
    }
    // 2) FiLM embedding MLP
    emb_mlp<<<B, 2 * CCH, 0, stream>>>(emb, emb_W, emb_b, ss);
    // 3) LN + SiLU -> h1 (f16, padded zero row)
    {
        int rowsPerBlk = 8, thr = 256;
        int blk = (N + 1 + rowsPerBlk - 1) / rowsPerBlk;
        ln1_silu<<<blk, thr, 0, stream>>>(feats, gamma, beta, h1, N);
    }
    // 4) sparse conv #1 (WMMA) -> c1o (f32)
    {
        int tiles = N / 16, wavesPerBlk = 8;
        spconv_wmma<false><<<(tiles + wavesPerBlk - 1) / wavesPerBlk, wavesPerBlk * 32, 0, stream>>>(
            h1, P1, b1, nidx, nullptr, c1o, N);
    }
    // 5) LN + FiLM + SiLU -> h2 (f16, padded zero row)
    {
        int rowsPerBlk = 8, thr = 256;
        int blk = (N + 1 + rowsPerBlk - 1) / rowsPerBlk;
        film_silu<<<blk, thr, 0, stream>>>(c1o, ss, bidx, h2, N);
    }
    // 6) sparse conv #2 (WMMA) + bias + residual -> d_out
    {
        int tiles = N / 16, wavesPerBlk = 8;
        spconv_wmma<true><<<(tiles + wavesPerBlk - 1) / wavesPerBlk, wavesPerBlk * 32, 0, stream>>>(
            h2, P2, b2, nidx, feats, outp, N);
    }
}
